// FlashAttention_15788299780241
// MI455X (gfx1250) — compile-verified, matrix operand order checked by reading, not run on hardware
//
#include <hip/hip_runtime.h>

// ---------------------------------------------------------------------------
// MHA forward for MI455X (gfx1250, wave32, WMMA 16x16x32 bf16)
// B=4, S=2048, D=2048, H=16, HEAD_DIM=128
// Data movement: TDM tensor_load_to_lds (K tiles), global_load_async_to_lds
// (GEMM A tiles), register-staged transpose for B/V tiles.
// ---------------------------------------------------------------------------

typedef __bf16 bf16;
typedef __attribute__((ext_vector_type(4)))  __bf16 bf16x4;
typedef __attribute__((ext_vector_type(8)))  __bf16 bf16x8;
typedef __attribute__((ext_vector_type(16))) __bf16 v16bf;
typedef __attribute__((ext_vector_type(8)))  float  v8f;
typedef __attribute__((ext_vector_type(4)))  float  f32x4;
typedef __attribute__((ext_vector_type(4)))  unsigned int u32x4;
typedef __attribute__((ext_vector_type(8)))  int    i32x8;
typedef __attribute__((ext_vector_type(4)))  int    i32x4;
typedef __attribute__((ext_vector_type(4)))  int    v4i;

#if __has_builtin(__builtin_amdgcn_global_load_async_to_lds_b128)
#define HAVE_ASYNC 1
#else
#define HAVE_ASYNC 0
#endif
#if __has_builtin(__builtin_amdgcn_tensor_load_to_lds)
#define HAVE_TDM 1
#else
#define HAVE_TDM 0
#endif

// global (AS1) / LDS (AS3) pointers with the v4i pointee the builtins expect
#define GVP(p) ((__attribute__((address_space(1))) v4i*)(p))
#define LVP(p) ((__attribute__((address_space(3))) v4i*)(p))

constexpr int   BB  = 4;
constexpr int   SS  = 2048;
constexpr int   DD  = 2048;
constexpr int   HH  = 16;
constexpr int   HD  = 128;
constexpr float ATT_SCALE = 0.08838834764831845f;  // 128^-0.5

__device__ __forceinline__ v16bf cat8(bf16x8 lo, bf16x8 hi) {
  return __builtin_shufflevector(lo, hi, 0,1,2,3,4,5,6,7,8,9,10,11,12,13,14,15);
}

__device__ __forceinline__ v8f wmma_bf16(v16bf a, v16bf b, v8f c) {
  // 8 args: (neg_a, A, neg_b, B, c_mod, C, reuse_a, reuse_b)
  return __builtin_amdgcn_wmma_f32_16x16x32_bf16(false, a, false, b, (short)0, c,
                                                 false, false);
}

// ---------------------------------------------------------------------------
// f32 -> bf16 conversion, 4-wide
// ---------------------------------------------------------------------------
__global__ __launch_bounds__(256)
void cvt4_kernel(const f32x4* __restrict__ in, bf16x4* __restrict__ out, long n4) {
  long i  = (long)blockIdx.x * blockDim.x + threadIdx.x;
  long st = (long)gridDim.x * blockDim.x;
  for (; i < n4; i += st) {
    f32x4 v = in[i];
    out[i] = __builtin_convertvector(v, bf16x4);
  }
}

// ---------------------------------------------------------------------------
// C[M,N] = A[M,K] * B[K,N] + bias   (A,B bf16, f32 accumulate)
// 128x128 workgroup tile, 8 waves in 4(M) x 2(N), wave tile 32x64 (2x4 wmma).
// A tile: async copy global->LDS (ASYNCcnt). B tile: register-staged
// transpose so B-fragments load as ds_load_b128. Double-buffered LDS.
// ---------------------------------------------------------------------------
template<bool OUT_BF16>
__global__ __launch_bounds__(256)
void gemm_bf16_kernel(const bf16* __restrict__ A, const bf16* __restrict__ Bm,
                      const float* __restrict__ bias, void* __restrict__ Cout,
                      int M, int N, int K)
{
  constexpr int PAD = 8;
  __shared__ alignas(16) bf16 sA [2][128][32 + PAD];  // [m][k]
  __shared__ alignas(16) bf16 sBt[2][128][32 + PAD];  // [n][k] (transposed)

  const int tid  = threadIdx.x;
  const int lane = tid & 31;
  const int w    = tid >> 5;
  const int waveM = w & 3;       // 0..3  -> 32-row group
  const int waveN = w >> 2;      // 0..1  -> 64-col group
  const int l16  = lane & 15;
  const int half = lane >> 4;

  const int m0 = blockIdx.y * 128;
  const int n0 = blockIdx.x * 128;

  v8f acc[2][4];
#pragma unroll
  for (int i = 0; i < 2; ++i)
#pragma unroll
    for (int j = 0; j < 4; ++j)
      acc[i][j] = (v8f){0.f,0.f,0.f,0.f,0.f,0.f,0.f,0.f};

  // per-thread cooperative-load chunk assignments (8 bf16 per chunk)
  // A tile: 128 rows x 32 k = 512 chunks ; B tile: 32 k x 128 n = 512 chunks
  const int ar0 = (tid)       >> 2, ak0 = ((tid)       & 3) * 8;
  const int ar1 = (tid + 256) >> 2, ak1 = ((tid + 256) & 3) * 8;
  const int bk0 = (tid)       >> 4, bn0 = ((tid)       & 15) * 8;
  const int bk1 = (tid + 256) >> 4, bn1 = ((tid + 256) & 15) * 8;

  const int steps = K / 32;

  // ---- preload step 0 into buffer 0 ----
#if HAVE_ASYNC
  __builtin_amdgcn_global_load_async_to_lds_b128(
      GVP(A + (size_t)(m0 + ar0) * K + ak0), LVP(&sA[0][ar0][ak0]), 0, 0);
  __builtin_amdgcn_global_load_async_to_lds_b128(
      GVP(A + (size_t)(m0 + ar1) * K + ak1), LVP(&sA[0][ar1][ak1]), 0, 0);
#else
  {
    bf16x8 va0 = *(const bf16x8*)(A + (size_t)(m0 + ar0) * K + ak0);
    bf16x8 va1 = *(const bf16x8*)(A + (size_t)(m0 + ar1) * K + ak1);
    *(bf16x8*)&sA[0][ar0][ak0] = va0;
    *(bf16x8*)&sA[0][ar1][ak1] = va1;
  }
#endif
  {
    bf16x8 vb0 = *(const bf16x8*)(Bm + (size_t)(bk0) * N + n0 + bn0);
    bf16x8 vb1 = *(const bf16x8*)(Bm + (size_t)(bk1) * N + n0 + bn1);
#pragma unroll
    for (int e = 0; e < 8; ++e) { sBt[0][bn0 + e][bk0] = vb0[e]; sBt[0][bn1 + e][bk1] = vb1[e]; }
  }
#if HAVE_ASYNC
  asm volatile("s_wait_asynccnt 0x0" ::: "memory");
#endif
  __syncthreads();

  for (int s = 0; s < steps; ++s) {
    const int buf = s & 1;
    const int nb  = buf ^ 1;
    const bool hasNext = (s + 1 < steps);

    bf16x8 pb0, pb1;
    if (hasNext) {
      const size_t k0 = (size_t)(s + 1) * 32;
#if HAVE_ASYNC
      // async A copy directly into next LDS buffer, overlapped with compute
      __builtin_amdgcn_global_load_async_to_lds_b128(
          GVP(A + (size_t)(m0 + ar0) * K + k0 + ak0), LVP(&sA[nb][ar0][ak0]), 0, 0);
      __builtin_amdgcn_global_load_async_to_lds_b128(
          GVP(A + (size_t)(m0 + ar1) * K + k0 + ak1), LVP(&sA[nb][ar1][ak1]), 0, 0);
#endif
      pb0 = *(const bf16x8*)(Bm + (k0 + bk0) * N + n0 + bn0);
      pb1 = *(const bf16x8*)(Bm + (k0 + bk1) * N + n0 + bn1);
    }

    // fragments + 8 WMMAs
    v16bf av[2], bv[4];
#pragma unroll
    for (int im = 0; im < 2; ++im) {
      const int row = waveM * 32 + im * 16 + l16;
      bf16x8 lo = *(const bf16x8*)&sA[buf][row][half * 8];
      bf16x8 hi = *(const bf16x8*)&sA[buf][row][16 + half * 8];
      av[im] = cat8(lo, hi);
    }
#pragma unroll
    for (int in = 0; in < 4; ++in) {
      const int col = waveN * 64 + in * 16 + l16;
      bf16x8 lo = *(const bf16x8*)&sBt[buf][col][half * 8];
      bf16x8 hi = *(const bf16x8*)&sBt[buf][col][16 + half * 8];
      bv[in] = cat8(lo, hi);
    }
#pragma unroll
    for (int im = 0; im < 2; ++im)
#pragma unroll
      for (int in = 0; in < 4; ++in)
        acc[im][in] = wmma_bf16(av[im], bv[in], acc[im][in]);

    __syncthreads();
    if (hasNext) {
#pragma unroll
      for (int e = 0; e < 8; ++e) { sBt[nb][bn0 + e][bk0] = pb0[e]; sBt[nb][bn1 + e][bk1] = pb1[e]; }
#if !HAVE_ASYNC
      const size_t k0 = (size_t)(s + 1) * 32;
      bf16x8 pa0 = *(const bf16x8*)(A + (size_t)(m0 + ar0) * K + k0 + ak0);
      bf16x8 pa1 = *(const bf16x8*)(A + (size_t)(m0 + ar1) * K + k0 + ak1);
      *(bf16x8*)&sA[nb][ar0][ak0] = pa0;
      *(bf16x8*)&sA[nb][ar1][ak1] = pa1;
#endif
    }
#if HAVE_ASYNC
    asm volatile("s_wait_asynccnt 0x0" ::: "memory");
#endif
    __syncthreads();
  }

  // epilogue: bias add + store (C/D layout: lane=col, vgpr r=row, half splits M by 8)
#pragma unroll
  for (int in = 0; in < 4; ++in) {
    const int n = n0 + waveN * 64 + in * 16 + l16;
    const float bsv = bias[n];
#pragma unroll
    for (int im = 0; im < 2; ++im) {
      const int mrow = m0 + waveM * 32 + im * 16 + half * 8;
#pragma unroll
      for (int r = 0; r < 8; ++r) {
        const float v = acc[im][in][r] + bsv;
        if (OUT_BF16) ((bf16*)Cout)[(size_t)(mrow + r) * N + n] = (bf16)v;
        else          ((float*)Cout)[(size_t)(mrow + r) * N + n] = v;
      }
    }
  }
}

// ---------------------------------------------------------------------------
// Flash attention: grid = (S/128, H, B), 256 threads = 8 waves.
// Wave w owns queries [q0 + 16w, q0 + 16w + 16). K-chunks of 64 keys.
// K tile fetched by the Tensor Data Mover (TENSORcnt); the TDM pad fields
// (interval 64 DW, amount 4 DW) reproduce the +8 bf16 LDS row padding.
// ---------------------------------------------------------------------------
__global__ __launch_bounds__(256)
void flash_attn_kernel(const bf16* __restrict__ Qp, const bf16* __restrict__ Kp,
                       const bf16* __restrict__ Vp, const int* __restrict__ maskp,
                       bf16* __restrict__ Op)
{
  constexpr int PK = 8;
  __shared__ alignas(16) bf16 sK [64][128 + PK];      // [key][d]
  __shared__ alignas(16) bf16 sVt[128][64 + PK];      // [d][key]   (transposed)
  __shared__ alignas(16) bf16 sP [8][16][64 + PK];    // per-wave [q][key]

  const int tid  = threadIdx.x;
  const int lane = tid & 31;
  const int w    = tid >> 5;
  const int l16  = lane & 15;
  const int half = lane >> 4;

  const int b  = blockIdx.z;
  const int h  = blockIdx.y;
  const int q0 = blockIdx.x * 128;
  const int qw = q0 + w * 16;

  const size_t headOff = (size_t)h * HD;
  const size_t batRow  = (size_t)b * SS;

  // Q fragments (A layout): 4 d-chunks of 32, held in registers for all chunks
  v16bf qf[4];
  {
    const bf16* qrow = Qp + (batRow + (size_t)(qw + l16)) * DD + headOff;
#pragma unroll
    for (int dc = 0; dc < 4; ++dc) {
      bf16x8 lo = *(const bf16x8*)(qrow + dc * 32 + half * 8);
      bf16x8 hi = *(const bf16x8*)(qrow + dc * 32 + 16 + half * 8);
      qf[dc] = cat8(lo, hi);
    }
  }

  v8f o[8];
#pragma unroll
  for (int dt = 0; dt < 8; ++dt) o[dt] = (v8f){0.f,0.f,0.f,0.f,0.f,0.f,0.f,0.f};
  float rmax[8], rsum[8];
#pragma unroll
  for (int r = 0; r < 8; ++r) { rmax[r] = -1e30f; rsum[r] = 0.f; }

  const int nChunks = SS / 64;
  for (int kc = 0; kc < nChunks; ++kc) {
    // ---- K chunk [64 keys x 128 d] via Tensor Data Mover ----
#if HAVE_TDM
    if (tid == 0) {
      const unsigned long long ga =
          (unsigned long long)(uintptr_t)(Kp + (batRow + (size_t)(kc * 64)) * DD + headOff);
      const unsigned ldsOff =
          (unsigned)(size_t)(__attribute__((address_space(3))) bf16*)&sK[0][0];
      // group0: count=1 | lds_addr | global_addr[56:0] | type=2
      u32x4 g0 = { 1u, ldsOff, (unsigned)ga,
                   (unsigned)((ga >> 32) & 0x01FFFFFFu) | 0x80000000u };
      // group1: data_size=2B, pad_enable, pad_interval=5(64DW), pad_amount=3(4DW),
      //         tensor_dim0=128, tensor_dim1=2048, tile0=128, tile1=64, stride0=2048
      i32x8 g1 = { 0x07510000, 0x00800000, 0x08000000, 0x00800000,
                   64, 2048, 0, 0 };
      i32x4 gz4 = { 0, 0, 0, 0 };
      i32x8 gz8 = { 0, 0, 0, 0, 0, 0, 0, 0 };
      __builtin_amdgcn_tensor_load_to_lds(g0, g1, gz4, gz4, gz8, 0);
    }
#else
#pragma unroll
    for (int i = 0; i < 4; ++i) {
      const int c    = tid + i * 256;
      const int krow = c >> 4;
      const int doff = (c & 15) * 8;
      bf16x8 kv = *(const bf16x8*)(Kp + (batRow + (size_t)(kc * 64 + krow)) * DD + headOff + doff);
      *(bf16x8*)&sK[krow][doff] = kv;
    }
#endif
    // ---- V chunk, transposed into LDS (register-staged) ----
#pragma unroll
    for (int i = 0; i < 4; ++i) {
      const int c    = tid + i * 256;
      const int krow = c >> 4;
      const int doff = (c & 15) * 8;
      bf16x8 vv = *(const bf16x8*)(Vp + (batRow + (size_t)(kc * 64 + krow)) * DD + headOff + doff);
#pragma unroll
      for (int e = 0; e < 8; ++e) sVt[doff + e][krow] = vv[e];
    }
#if HAVE_TDM
    asm volatile("s_wait_tensorcnt 0x0" ::: "memory");
#endif
    __syncthreads();

    // ---- scores: S[16q x 64k] = Q * K^T  (16 WMMAs) ----
    v8f sc[4];
#pragma unroll
    for (int nt = 0; nt < 4; ++nt) sc[nt] = (v8f){0.f,0.f,0.f,0.f,0.f,0.f,0.f,0.f};
#pragma unroll
    for (int dc = 0; dc < 4; ++dc) {
#pragma unroll
      for (int nt = 0; nt < 4; ++nt) {
        const int key = nt * 16 + l16;
        bf16x8 lo = *(const bf16x8*)&sK[key][dc * 32 + half * 8];
        bf16x8 hi = *(const bf16x8*)&sK[key][dc * 32 + 16 + half * 8];
        sc[nt] = wmma_bf16(qf[dc], cat8(lo, hi), sc[nt]);
      }
    }

    // ---- scale + mask ----
#pragma unroll
    for (int nt = 0; nt < 4; ++nt) {
#pragma unroll
      for (int r = 0; r < 8; ++r) {
        float v = sc[nt][r] * ATT_SCALE;
        const int q = qw + half * 8 + r;
        const int k = kc * 64 + nt * 16 + l16;
        if (maskp[(size_t)b * SS * SS + (size_t)q * SS + k] == 0) v = -3.0e38f;
        sc[nt][r] = v;
      }
    }

    // ---- online softmax (rows live in (half, r); reduce across 16 lanes) ----
    float cmax[8];
#pragma unroll
    for (int r = 0; r < 8; ++r) {
      float m = sc[0][r];
#pragma unroll
      for (int nt = 1; nt < 4; ++nt) m = fmaxf(m, sc[nt][r]);
      cmax[r] = m;
    }
#pragma unroll
    for (int sh = 1; sh < 16; sh <<= 1)
#pragma unroll
      for (int r = 0; r < 8; ++r)
        cmax[r] = fmaxf(cmax[r], __shfl_xor(cmax[r], sh, 32));

    float corr[8];
#pragma unroll
    for (int r = 0; r < 8; ++r) {
      const float mn = fmaxf(rmax[r], cmax[r]);
      corr[r] = __expf(rmax[r] - mn);
      rmax[r] = mn;
    }

    float csum[8];
#pragma unroll
    for (int r = 0; r < 8; ++r) csum[r] = 0.f;
#pragma unroll
    for (int nt = 0; nt < 4; ++nt) {
#pragma unroll
      for (int r = 0; r < 8; ++r) {
        const float p = __expf(sc[nt][r] - rmax[r]);
        csum[r] += p;
        sP[w][half * 8 + r][nt * 16 + l16] = (bf16)p;
      }
    }
#pragma unroll
    for (int sh = 1; sh < 16; sh <<= 1)
#pragma unroll
      for (int r = 0; r < 8; ++r)
        csum[r] += __shfl_xor(csum[r], sh, 32);
#pragma unroll
    for (int r = 0; r < 8; ++r) rsum[r] = rsum[r] * corr[r] + csum[r];
#pragma unroll
    for (int dt = 0; dt < 8; ++dt)
#pragma unroll
      for (int r = 0; r < 8; ++r) o[dt][r] *= corr[r];

    // ---- O += P * V  (16 WMMAs) ----
#pragma unroll
    for (int kk = 0; kk < 2; ++kk) {
      bf16x8 plo = *(const bf16x8*)&sP[w][l16][kk * 32 + half * 8];
      bf16x8 phi = *(const bf16x8*)&sP[w][l16][kk * 32 + 16 + half * 8];
      v16bf pf = cat8(plo, phi);
#pragma unroll
      for (int dt = 0; dt < 8; ++dt) {
        const int d = dt * 16 + l16;
        bf16x8 vlo = *(const bf16x8*)&sVt[d][kk * 32 + half * 8];
        bf16x8 vhi = *(const bf16x8*)&sVt[d][kk * 32 + 16 + half * 8];
        o[dt] = wmma_bf16(pf, cat8(vlo, vhi), o[dt]);
      }
    }
    __syncthreads();
  }

  // ---- epilogue: normalize, store bf16 attention output in [B*S, D] layout ----
#pragma unroll
  for (int r = 0; r < 8; ++r) rsum[r] = 1.0f / rsum[r];
#pragma unroll
  for (int dt = 0; dt < 8; ++dt) {
    const size_t d = headOff + dt * 16 + l16;
#pragma unroll
    for (int r = 0; r < 8; ++r) {
      const int q = qw + half * 8 + r;
      Op[(batRow + q) * DD + d] = (bf16)(o[dt][r] * rsum[r]);
    }
  }
}

// ---------------------------------------------------------------------------
// Launcher
// ---------------------------------------------------------------------------
extern "C" void kernel_launch(void* const* d_in, const int* in_sizes, int n_in,
                              void* d_out, int out_size, void* d_ws, size_t ws_size,
                              hipStream_t stream) {
  const float* x    = (const float*)d_in[0];
  const int*   mask = (const int*)  d_in[1];
  const float* Wq   = (const float*)d_in[2];
  const float* bq   = (const float*)d_in[3];
  const float* Wk   = (const float*)d_in[4];
  const float* bk   = (const float*)d_in[5];
  const float* Wv   = (const float*)d_in[6];
  const float* bv   = (const float*)d_in[7];
  const float* Wo   = (const float*)d_in[8];
  const float* bo   = (const float*)d_in[9];
  float* out = (float*)d_out;

  const size_t nx = (size_t)BB * SS * DD;   // 16,777,216
  const size_t nw = (size_t)DD * DD;        //  4,194,304

  bf16* p   = (bf16*)d_ws;
  bf16* xb  = p; p += nx;
  bf16* Wqb = p; p += nw;
  bf16* Wkb = p; p += nw;
  bf16* Wvb = p; p += nw;
  bf16* Wob = p; p += nw;
  bf16* Qb  = p; p += nx;
  bf16* Kb  = p; p += nx;
  bf16* Vb  = p; p += nx;
  bf16* Ob  = p; p += nx;

  // 1) f32 -> bf16 conversions
  cvt4_kernel<<<dim3((unsigned)(nx / 4 / 256)), 256, 0, stream>>>((const f32x4*)x,  (bf16x4*)xb,  (long)(nx / 4));
  cvt4_kernel<<<dim3((unsigned)(nw / 4 / 256)), 256, 0, stream>>>((const f32x4*)Wq, (bf16x4*)Wqb, (long)(nw / 4));
  cvt4_kernel<<<dim3((unsigned)(nw / 4 / 256)), 256, 0, stream>>>((const f32x4*)Wk, (bf16x4*)Wkb, (long)(nw / 4));
  cvt4_kernel<<<dim3((unsigned)(nw / 4 / 256)), 256, 0, stream>>>((const f32x4*)Wv, (bf16x4*)Wvb, (long)(nw / 4));
  cvt4_kernel<<<dim3((unsigned)(nw / 4 / 256)), 256, 0, stream>>>((const f32x4*)Wo, (bf16x4*)Wob, (long)(nw / 4));

  // 2) Q/K/V projections: [8192,2048] x [2048,2048] + bias -> bf16
  const int M = BB * SS, N = DD, K = DD;
  dim3 ggrid(N / 128, M / 128);
  gemm_bf16_kernel<true><<<ggrid, 256, 0, stream>>>(xb, Wqb, bq, (void*)Qb, M, N, K);
  gemm_bf16_kernel<true><<<ggrid, 256, 0, stream>>>(xb, Wkb, bk, (void*)Kb, M, N, K);
  gemm_bf16_kernel<true><<<ggrid, 256, 0, stream>>>(xb, Wvb, bv, (void*)Vb, M, N, K);

  // 3) flash attention
  flash_attn_kernel<<<dim3(SS / 128, HH, BB), 256, 0, stream>>>(Qb, Kb, Vb, mask, Ob);

  // 4) output projection -> f32 d_out
  gemm_bf16_kernel<false><<<ggrid, 256, 0, stream>>>(Ob, Wob, bo, (void*)out, M, N, K);
}
